// MAB_15659450761353
// MI455X (gfx1250) — matrix operand order repaired and checked by reading, not compiled, not run on hardware
//
#include <hip/hip_runtime.h>

// ---------------------------------------------------------------------------
// Feature probe: gfx1250 async global->LDS copy
// ---------------------------------------------------------------------------
#if defined(__has_builtin)
#if __has_builtin(__builtin_amdgcn_global_load_async_to_lds_b128)
#define HAVE_ASYNC_LDS 1
#endif
#endif
#ifndef HAVE_ASYNC_LDS
#define HAVE_ASYNC_LDS 0
#endif

// ---------------------------------------------------------------------------
// Types / helpers
// ---------------------------------------------------------------------------
typedef __attribute__((ext_vector_type(16))) __bf16 bf16x16;
typedef __attribute__((ext_vector_type(8)))  float  f32x8;

#if HAVE_ASYNC_LDS
// Builtin signature (from hipcc diagnostic): param0 is
// '__attribute__((__vector_size__(4*sizeof(int)))) int __device__ *', i.e. an
// int4 vector pointer in address_space(1); LDS side is address_space(3).
typedef int v4i __attribute__((vector_size(16)));
typedef __attribute__((address_space(1))) v4i gbl_v4i;
typedef __attribute__((address_space(3))) v4i lds_v4i;
#endif

union FragU {
    bf16x16 v;
    uint4   u[2];
};

__device__ inline f32x8 zero8() {
    f32x8 z;
#pragma unroll
    for (int i = 0; i < 8; ++i) z[i] = 0.0f;
    return z;
}

__device__ inline unsigned short f2bf(float f) {
    unsigned x = __float_as_uint(f);
    unsigned r = x + 0x7fffu + ((x >> 16) & 1u);   // round-to-nearest-even
    return (unsigned short)(r >> 16);
}

__device__ inline f32x8 wmma_bf16(const bf16x16& a, const bf16x16& b, const f32x8& c) {
    // (neg_a, A, neg_b, B, c_mod, C, reuse_a, reuse_b)
    return __builtin_amdgcn_wmma_f32_16x16x32_bf16(false, a, false, b, (short)0, c,
                                                   false, false);
}

// 16-byte global->LDS copy: async DMA when available, VGPR round-trip otherwise.
__device__ inline void copy16_to_lds(const unsigned short* g, unsigned short* l) {
#if HAVE_ASYNC_LDS
    __builtin_amdgcn_global_load_async_to_lds_b128((gbl_v4i*)g, (lds_v4i*)l,
                                                   0, 0);
#else
    *(uint4*)l = *(const uint4*)g;
#endif
}

__device__ inline void wait_async_copies() {
#if HAVE_ASYNC_LDS
    asm volatile("s_wait_asynccnt 0x0" ::: "memory");
#endif
}

// Problem constants (match reference)
#define NQ_  2048
#define NK_  2048
#define DD_  512
#define HH_  8
#define DH_  64
#define MM_  (2 * NQ_)   // B*NQ = 4096 flattened rows

// ---------------------------------------------------------------------------
// fp32 -> bf16 conversion
// ---------------------------------------------------------------------------
__global__ __launch_bounds__(256) void cvt_bf16_kernel(const float* __restrict__ in,
                                                       unsigned short* __restrict__ out,
                                                       int n) {
    int i = blockIdx.x * 256 + threadIdx.x;
    if (i < n) out[i] = f2bf(in[i]);
}

// ---------------------------------------------------------------------------
// Generic GEMM: out[M,N] = A[M,K] * W[N,K]^T  (both K-contiguous, bf16 in,
// f32 accumulate). Epilogue: +bias, relu, +f32 residual, emit f32 and/or bf16.
// Block tile 128x64, 8 waves (4m x 2n) of 32x32, K stepped by 32 through
// DOUBLE-BUFFERED LDS with async global->LDS prefetch of the next chunk.
// ---------------------------------------------------------------------------
__global__ __launch_bounds__(256) void gemm_bf16_kernel(
    const unsigned short* __restrict__ A, const unsigned short* __restrict__ W,
    int M, int N, int K,
    const float* __restrict__ bias, const float* __restrict__ resid,
    float* __restrict__ outF, unsigned short* __restrict__ outB, int relu) {

    __shared__ unsigned short lA[2][128 * 32];  // 2 x 8 KB
    __shared__ unsigned short lW[2][64 * 32];   // 2 x 4 KB

    const int tid  = threadIdx.x;
    const int lane = tid & 31;
    const int w    = tid >> 5;       // wave 0..7
    const int wm   = w & 3;          // 4 waves along M
    const int wn   = w >> 2;         // 2 waves along N
    const int h4   = lane >> 4;      // lane half
    const int ln16 = lane & 15;
    const int m0   = blockIdx.x * 128;
    const int n0   = blockIdx.y * 64;

    // Stage one K-chunk (A tile 128x32 = 512 uint4, W tile 64x32 = 256 uint4)
    auto stage = [&](int kc, int buf) {
#pragma unroll
        for (int i = tid; i < 128 * 4; i += 256) {
            int r = i >> 2, c = (i & 3) * 8;
            copy16_to_lds(A + (size_t)(m0 + r) * K + kc + c, &lA[buf][i * 8]);
        }
        {
            int i = tid;              // exactly 256 uint4
            int r = i >> 2, c = (i & 3) * 8;
            copy16_to_lds(W + (size_t)(n0 + r) * K + kc + c, &lW[buf][i * 8]);
        }
    };

    f32x8 acc[2][2];
#pragma unroll
    for (int mt = 0; mt < 2; ++mt)
#pragma unroll
        for (int nt = 0; nt < 2; ++nt) acc[mt][nt] = zero8();

    stage(0, 0);
    wait_async_copies();
    __syncthreads();

    int cur = 0;
    for (int kc = 0; kc < K; kc += 32) {
        // Prefetch next chunk into the other buffer while we compute this one.
        if (kc + 32 < K) stage(kc + 32, cur ^ 1);

        FragU af[2], bf[2];
#pragma unroll
        for (int mt = 0; mt < 2; ++mt) {
            const unsigned short* p = &lA[cur][(wm * 32 + mt * 16 + ln16) * 32];
            af[mt].u[0] = *(const uint4*)(p + h4 * 8);
            af[mt].u[1] = *(const uint4*)(p + 16 + h4 * 8);
        }
#pragma unroll
        for (int nt = 0; nt < 2; ++nt) {
            const unsigned short* p = &lW[cur][(wn * 32 + nt * 16 + ln16) * 32];
            bf[nt].u[0] = *(const uint4*)(p + h4 * 8);
            bf[nt].u[1] = *(const uint4*)(p + 16 + h4 * 8);
        }
#pragma unroll
        for (int mt = 0; mt < 2; ++mt)
#pragma unroll
            for (int nt = 0; nt < 2; ++nt)
                acc[mt][nt] = wmma_bf16(af[mt].v, bf[nt].v, acc[mt][nt]);

        // Next-chunk copies must have landed before anyone reads them; the
        // barrier also fences re-use of the buffer we just computed from.
        wait_async_copies();
        __syncthreads();
        cur ^= 1;
    }

    // Epilogue. C layout: element v -> row (v + 8*h4), col ln16.
#pragma unroll
    for (int mt = 0; mt < 2; ++mt)
#pragma unroll
        for (int nt = 0; nt < 2; ++nt)
#pragma unroll
            for (int v = 0; v < 8; ++v) {
                int row = m0 + wm * 32 + mt * 16 + v + 8 * h4;
                int col = n0 + wn * 32 + nt * 16 + ln16;
                float x = acc[mt][nt][v];
                if (bias)  x += bias[col];
                if (relu)  x = fmaxf(x, 0.0f);
                if (resid) x += resid[(size_t)row * N + col];
                size_t idx = (size_t)row * N + col;
                if (outF) outF[idx] = x;
                if (outB) outB[idx] = f2bf(x);
            }
}

// ---------------------------------------------------------------------------
// Flash attention. One block = (b, h, 128 q rows); 8 waves x 16 q rows.
// Streams NK in chunks of 32 keys. Matches reference semantics:
//   masked logits -> clip(-inf) == -30 (!), all-masked rows -> 0.
// ---------------------------------------------------------------------------
__global__ __launch_bounds__(256) void attn_kernel(
    const unsigned short* __restrict__ Qbf,   // [B*NQ, 512], col = h*64+d
    const unsigned short* __restrict__ Kbf,   // [B*NK, 512]
    const unsigned short* __restrict__ Vbf,   // [B*NK, 512]
    const unsigned char* __restrict__ maskK,  // [B, NK] bool
    unsigned short* __restrict__ Xbf) {       // [B*NQ, 512]

    __shared__ unsigned short lVt[64 * 40];   // V tile transposed [d][key], 5 KB
    __shared__ unsigned short lP[8][16 * 32]; // per-wave P tiles, 8 KB

    const int tid  = threadIdx.x;
    const int lane = tid & 31;
    const int w    = tid >> 5;
    const int h4   = lane >> 4;
    const int ln16 = lane & 15;

    const int blk = blockIdx.x;           // 256 blocks
    const int qb  = blk & 15;
    const int h   = (blk >> 4) & 7;
    const int b   = blk >> 7;
    const int qrow0 = qb * 128 + w * 16;

    // Q fragments (A layout) straight from global: rows = qrow0 + ln16
    FragU qf[2];
    {
        const unsigned short* qp =
            Qbf + (size_t)(b * NQ_ + qrow0 + ln16) * DD_ + h * DH_;
#pragma unroll
        for (int c = 0; c < 2; ++c) {
            qf[c].u[0] = *(const uint4*)(qp + c * 32 + h4 * 8);
            qf[c].u[1] = *(const uint4*)(qp + c * 32 + 16 + h4 * 8);
        }
    }

    f32x8 O[4];
#pragma unroll
    for (int t = 0; t < 4; ++t) O[t] = zero8();
    float mrow[8], lrow[8];
#pragma unroll
    for (int v = 0; v < 8; ++v) { mrow[v] = -1e30f; lrow[v] = 0.0f; }
    int hasAny = 0;
    const float scale = 0.125f;  // DH^-0.5

    for (int kc = 0; kc < NK_; kc += 32) {
        __syncthreads();
        // Load V chunk transposed into LDS: [d][key], row stride 40
        {
            int key = tid >> 3;           // 0..31
            int d0  = (tid & 7) * 8;      // 0..56
            union { uint4 q; unsigned short s[8]; } vv;
            vv.q = *(const uint4*)(Vbf + (size_t)(b * NK_ + kc + key) * DD_ +
                                   h * DH_ + d0);
#pragma unroll
            for (int j = 0; j < 8; ++j) lVt[(d0 + j) * 40 + key] = vv.s[j];
        }
        __syncthreads();

        // ---- logits for the two 16-key sub-tiles ----
        float s_[2][8];
        int anyv = 0;
#pragma unroll
        for (int kt = 0; kt < 2; ++kt) {
            FragU kf0, kf1;
            const unsigned short* kp =
                Kbf + (size_t)(b * NK_ + kc + kt * 16 + ln16) * DD_ + h * DH_;
            kf0.u[0] = *(const uint4*)(kp + h4 * 8);
            kf0.u[1] = *(const uint4*)(kp + 16 + h4 * 8);
            kf1.u[0] = *(const uint4*)(kp + 32 + h4 * 8);
            kf1.u[1] = *(const uint4*)(kp + 32 + 16 + h4 * 8);
            f32x8 sacc = zero8();
            sacc = wmma_bf16(qf[0].v, kf0.v, sacc);
            sacc = wmma_bf16(qf[1].v, kf1.v, sacc);
            int col   = kc + kt * 16 + ln16;
            int valid = maskK[b * NK_ + col] != 0;
            anyv |= valid;
#pragma unroll
            for (int v = 0; v < 8; ++v) {
                // Reference: where(mask,logit,-inf) then clip(-30,30):
                // masked entries become exactly -30.
                float s = valid ? fminf(fmaxf(sacc[v] * scale, -30.0f), 30.0f)
                                : -30.0f;
                s_[kt][v] = s;
            }
        }
        hasAny |= __any(anyv);

        // ---- online softmax update ----
        float alpha[8];
#pragma unroll
        for (int v = 0; v < 8; ++v) {
            float mx = fmaxf(s_[0][v], s_[1][v]);
#pragma unroll
            for (int off = 1; off < 16; off <<= 1)
                mx = fmaxf(mx, __shfl_xor(mx, off, 32));
            float nm = fmaxf(mrow[v], mx);
            float a  = __expf(mrow[v] - nm);
            float p0 = __expf(s_[0][v] - nm);
            float p1 = __expf(s_[1][v] - nm);
            float rs = p0 + p1;
#pragma unroll
            for (int off = 1; off < 16; off <<= 1)
                rs += __shfl_xor(rs, off, 32);
            lrow[v]  = lrow[v] * a + rs;
            mrow[v]  = nm;
            alpha[v] = a;
            s_[0][v] = p0;
            s_[1][v] = p1;
        }
#pragma unroll
        for (int t = 0; t < 4; ++t)
#pragma unroll
            for (int v = 0; v < 8; ++v) O[t][v] *= alpha[v];

        // ---- re-layout P (C layout -> A layout) through per-wave LDS ----
#pragma unroll
        for (int kt = 0; kt < 2; ++kt)
#pragma unroll
            for (int v = 0; v < 8; ++v)
                lP[w][(v + 8 * h4) * 32 + kt * 16 + ln16] = f2bf(s_[kt][v]);
        // Same-wave LDS RAW: wait for DS writes before the vector reads.
        asm volatile("s_wait_dscnt 0x0" ::: "memory");
        FragU pf;
        {
            const unsigned short* pp = &lP[w][ln16 * 32];
            pf.u[0] = *(const uint4*)(pp + h4 * 8);
            pf.u[1] = *(const uint4*)(pp + 16 + h4 * 8);
        }

        // ---- O += P * V  (4 n-tiles of 16 d-columns) ----
#pragma unroll
        for (int nt = 0; nt < 4; ++nt) {
            FragU vf;
            const unsigned short* vp = &lVt[(nt * 16 + ln16) * 40];
            vf.u[0] = *(const uint4*)(vp + h4 * 8);
            vf.u[1] = *(const uint4*)(vp + 16 + h4 * 8);
            O[nt] = wmma_bf16(pf.v, vf.v, O[nt]);
        }
    }

    // ---- finalize: divide by row sums; all-masked rows -> 0 ----
#pragma unroll
    for (int nt = 0; nt < 4; ++nt)
#pragma unroll
        for (int v = 0; v < 8; ++v) {
            int row = qrow0 + v + 8 * h4;
            int d   = nt * 16 + ln16;
            float x = (hasAny && lrow[v] > 0.0f) ? O[nt][v] / lrow[v] : 0.0f;
            Xbf[(size_t)(b * NQ_ + row) * DD_ + h * DH_ + d] = f2bf(x);
        }
}

// ---------------------------------------------------------------------------
// Row-wise LayerNorm (D=512). One block (128 threads) per row.
// ---------------------------------------------------------------------------
__global__ __launch_bounds__(128) void ln_kernel(
    const float* __restrict__ X, const float* __restrict__ g,
    const float* __restrict__ be, float* __restrict__ outF,
    unsigned short* __restrict__ outB, int D) {

    const int row = blockIdx.x;
    const float* x = X + (size_t)row * D;
    float s = 0.0f, s2 = 0.0f;
    for (int i = threadIdx.x; i < D; i += 128) {
        float v = x[i];
        s += v;
        s2 += v * v;
    }
    __shared__ float rs[128], rs2[128];
    rs[threadIdx.x] = s;
    rs2[threadIdx.x] = s2;
    __syncthreads();
    for (int off = 64; off > 0; off >>= 1) {
        if (threadIdx.x < off) {
            rs[threadIdx.x] += rs[threadIdx.x + off];
            rs2[threadIdx.x] += rs2[threadIdx.x + off];
        }
        __syncthreads();
    }
    float mean = rs[0] / (float)D;
    float var  = rs2[0] / (float)D - mean * mean;
    float inv  = rsqrtf(var + 1e-4f);
    for (int i = threadIdx.x; i < D; i += 128) {
        float y = (x[i] - mean) * inv * g[i] + be[i];
        if (outF) outF[(size_t)row * D + i] = y;
        if (outB) outB[(size_t)row * D + i] = f2bf(y);
    }
}

// ---------------------------------------------------------------------------
// Orchestration
// ---------------------------------------------------------------------------
extern "C" void kernel_launch(void* const* d_in, const int* in_sizes, int n_in,
                              void* d_out, int out_size, void* d_ws,
                              size_t ws_size, hipStream_t stream) {
    (void)in_sizes; (void)n_in; (void)out_size; (void)ws_size;

    const float* Q   = (const float*)d_in[0];
    const float* K   = (const float*)d_in[1];
    const float* Wq  = (const float*)d_in[2];
    const float* Wk  = (const float*)d_in[3];
    const float* Wv  = (const float*)d_in[4];
    const float* Wo  = (const float*)d_in[5];
    const float* W1  = (const float*)d_in[6];
    const float* b1  = (const float*)d_in[7];
    const float* W2  = (const float*)d_in[8];
    const float* b2  = (const float*)d_in[9];
    const float* g0  = (const float*)d_in[10];
    const float* be0 = (const float*)d_in[11];
    const float* g1  = (const float*)d_in[12];
    const float* be1 = (const float*)d_in[13];
    const unsigned char* maskK = (const unsigned char*)d_in[14];
    float* out = (float*)d_out;

    const int M = MM_;      // 4096 flattened rows
    const int D = DD_;      // 512

    size_t off = 0;
    auto alloc = [&](size_t bytes) -> void* {
        void* p = (char*)d_ws + off;
        off += (bytes + 255) & ~(size_t)255;
        return p;
    };
    typedef unsigned short u16;
    u16*   Qbf    = (u16*)alloc((size_t)M * D * 2);
    u16*   Kbf    = (u16*)alloc((size_t)M * D * 2);
    u16*   Wqbf   = (u16*)alloc((size_t)D * D * 2);
    u16*   Wkbf   = (u16*)alloc((size_t)D * D * 2);
    u16*   Wvbf   = (u16*)alloc((size_t)D * D * 2);
    u16*   Wobf   = (u16*)alloc((size_t)D * D * 2);
    u16*   W1bf   = (u16*)alloc((size_t)D * D * 2);
    u16*   W2bf   = (u16*)alloc((size_t)D * D * 2);
    float* qprojF = (float*)alloc((size_t)M * D * 4);
    u16*   qhB    = (u16*)alloc((size_t)M * D * 2);
    u16*   khB    = (u16*)alloc((size_t)M * D * 2);
    u16*   vhB    = (u16*)alloc((size_t)M * D * 2);
    u16*   XB     = (u16*)alloc((size_t)M * D * 2);
    float* h0pre  = (float*)alloc((size_t)M * D * 4);
    float* H0F    = (float*)alloc((size_t)M * D * 4);
    u16*   H0B    = (u16*)alloc((size_t)M * D * 2);
    u16*   y1B    = (u16*)alloc((size_t)M * D * 2);
    float* y2F    = (float*)alloc((size_t)M * D * 4);

    // 1) fp32 -> bf16 conversions
    int nAct = M * D;       // 2,097,152
    int nWgt = D * D;       // 262,144
    cvt_bf16_kernel<<<(nAct + 255) / 256, 256, 0, stream>>>(Q, Qbf, nAct);
    cvt_bf16_kernel<<<(nAct + 255) / 256, 256, 0, stream>>>(K, Kbf, nAct);
    cvt_bf16_kernel<<<(nWgt + 255) / 256, 256, 0, stream>>>(Wq, Wqbf, nWgt);
    cvt_bf16_kernel<<<(nWgt + 255) / 256, 256, 0, stream>>>(Wk, Wkbf, nWgt);
    cvt_bf16_kernel<<<(nWgt + 255) / 256, 256, 0, stream>>>(Wv, Wvbf, nWgt);
    cvt_bf16_kernel<<<(nWgt + 255) / 256, 256, 0, stream>>>(Wo, Wobf, nWgt);
    cvt_bf16_kernel<<<(nWgt + 255) / 256, 256, 0, stream>>>(W1, W1bf, nWgt);
    cvt_bf16_kernel<<<(nWgt + 255) / 256, 256, 0, stream>>>(W2, W2bf, nWgt);

    dim3 gg(M / 128, D / 64);   // 32 x 8 blocks

    // 2) projections: q_proj (f32 + bf16), K/V heads (bf16 only)
    gemm_bf16_kernel<<<gg, 256, 0, stream>>>(Qbf, Wqbf, M, D, D, nullptr,
                                             nullptr, qprojF, qhB, 0);
    gemm_bf16_kernel<<<gg, 256, 0, stream>>>(Kbf, Wkbf, M, D, D, nullptr,
                                             nullptr, nullptr, khB, 0);
    gemm_bf16_kernel<<<gg, 256, 0, stream>>>(Kbf, Wvbf, M, D, D, nullptr,
                                             nullptr, nullptr, vhB, 0);

    // 3) attention: 2*8*16 = 256 blocks
    attn_kernel<<<256, 256, 0, stream>>>(qhB, khB, vhB, maskK, XB);

    // 4) output proj + residual(q_proj), then LN0 -> H0 (f32 + bf16)
    gemm_bf16_kernel<<<gg, 256, 0, stream>>>(XB, Wobf, M, D, D, nullptr,
                                             qprojF, h0pre, nullptr, 0);
    ln_kernel<<<M, 128, 0, stream>>>(h0pre, g0, be0, H0F, H0B, D);

    // 5) FFN: relu(H0 W1^T + b1) W2^T + b2 + H0, then LN1 -> out
    gemm_bf16_kernel<<<gg, 256, 0, stream>>>(H0B, W1bf, M, D, D, b1, nullptr,
                                             nullptr, y1B, 1);
    gemm_bf16_kernel<<<gg, 256, 0, stream>>>(y1B, W2bf, M, D, D, b2, H0F, y2F,
                                             nullptr, 0);
    ln_kernel<<<M, 128, 0, stream>>>(y2F, g1, be1, out, nullptr, D);
}
